// GConvLSTM_32847909879986
// MI455X (gfx1250) — matrix operand order repaired and checked by reading, not compiled, not run on hardware
//
#include <hip/hip_runtime.h>

// ---------------------------------------------------------------------------
// GConvLSTM for MI455X (gfx1250, wave32, WMMA).
//   Per step:
//     A[:,0:128]   = X[t] @ W_pool^T                     (WMMA, bf16 out)
//     A[:,128:384] = [prop(Xp) | 2*prop(prop(Xp)) - Xp]  (bf16 basis, x side)
//     A[:,384:768] = [H | prop(H) | 2*prop(prop(H)) - H] (bf16 basis, h side)
//     Fused: G = A(Nx768) @ Bt(512x768) + bias  -> LDS -> LSTM gates
//            -> Cst, H, out[:,t,:]   (no global G buffer at all)
// ---------------------------------------------------------------------------

typedef __bf16 bf16_t;
typedef __attribute__((ext_vector_type(16))) __bf16 v16bf;
typedef __attribute__((ext_vector_type(8)))  __bf16 v8bf;
typedef __attribute__((ext_vector_type(4)))  __bf16 v4bf;
typedef __attribute__((ext_vector_type(8)))  float  v8f;
typedef __attribute__((ext_vector_type(8)))  float  f32x8;

#define T_LEN 4
#define FEAT  128
#define CH    128
#define KD    768   // 2 sides * 3 Chebyshev * 128
#define NG    512   // 4 gates * 128

__device__ __forceinline__ float sigf(float x) { return 1.0f / (1.0f + __expf(-x)); }

// ---------------------------------------------------------------- graph prep
__global__ void k_deg(const int* __restrict__ src, float* __restrict__ deg, int E) {
    int e = blockIdx.x * blockDim.x + threadIdx.x;
    if (e < E) atomicAdd(&deg[src[e]], 1.0f);
}
__global__ void k_dinv(const float* __restrict__ deg, float* __restrict__ dinv, int N) {
    int n = blockIdx.x * blockDim.x + threadIdx.x;
    if (n < N) { float d = deg[n]; dinv[n] = d > 0.0f ? rsqrtf(d) : 0.0f; }
}
__global__ void k_norm(const int* __restrict__ src, const int* __restrict__ dst,
                       const float* __restrict__ dinv, float* __restrict__ norm, int E) {
    int e = blockIdx.x * blockDim.x + threadIdx.x;
    if (e < E) norm[e] = -(dinv[src[e]] * dinv[dst[e]]);
}

// ------------------------------------------------------------- weight prep
// Combined column-major B: Bt[col*768 + slot*128 + r]
//   slot 0..2 -> conv_W[2g][kc][r][col] (x side), slot 3..5 -> conv_W[2g+1] (h side)
__global__ void k_prep_w(const float* __restrict__ convW, bf16_t* __restrict__ Bt) {
    int tid = blockIdx.x * blockDim.x + threadIdx.x;          // 4*3*128*128
    if (tid >= 4 * 3 * 128 * 128) return;
    int g   = tid / (3 * 16384);
    int rem = tid % (3 * 16384);
    int kc  = rem / 16384;
    int r   = (rem % 16384) / 128;
    int c   = rem % 128;
    size_t base = (size_t)(g * 128 + c) * KD + r;
    Bt[base + (size_t)kc * 128]       =
        (bf16_t)convW[(((size_t)(2 * g)     * 3 + kc) * 128 + r) * 128 + c];
    Bt[base + (size_t)(3 + kc) * 128] =
        (bf16_t)convW[(((size_t)(2 * g + 1) * 3 + kc) * 128 + r) * 128 + c];
}
__global__ void k_bias(const float* __restrict__ conv_b, float* __restrict__ bias512) {
    int tid = blockIdx.x * blockDim.x + threadIdx.x;
    if (tid >= NG) return;
    int g = tid >> 7, c = tid & 127;
    bias512[tid] = conv_b[(2 * g) * 128 + c] + conv_b[(2 * g + 1) * 128 + c];
}
__global__ void k_poolw(const float* __restrict__ Wp, bf16_t* __restrict__ Bp) {
    int tid = blockIdx.x * blockDim.x + threadIdx.x;
    if (tid < 128 * 128) Bp[tid] = (bf16_t)Wp[tid];   // W_pool[c][k] is already col-major B
}

// -------------------------------------------------------------- conversions
__global__ void k_cvt_slice(const float* __restrict__ s, bf16_t* __restrict__ A,
                            int coloff, int n) {
    int tid = blockIdx.x * blockDim.x + threadIdx.x;
    if (tid >= n) return;
    int row = tid >> 7, c = tid & 127;
    A[(size_t)row * KD + coloff + c] = (bf16_t)s[tid];
}
// Tx2 = 2*prop(Tx1) - Tx0  (Tx0 read back from bf16 slice `base`)
__global__ void k_tx2(const float* __restrict__ P, bf16_t* __restrict__ A,
                      int base, int n) {
    int tid = blockIdx.x * blockDim.x + threadIdx.x;
    if (tid >= n) return;
    int row = tid >> 7, c = tid & 127;
    size_t b = (size_t)row * KD + base;
    A[b + 256 + c] = (bf16_t)(2.0f * P[tid] - (float)A[b + c]);
}

// ----------------------------------------------------- sparse propagation
// out[dst] += norm[e] * X[src]; bf16 gather (halves gather bytes), fp32 scatter.
// Grid-stride over edges with prefetch of the next gather row (global_prefetch_b8).
__global__ void k_prop(const bf16_t* __restrict__ X, int coloff,
                       const int* __restrict__ src, const int* __restrict__ dst,
                       const float* __restrict__ norm, float* __restrict__ outp, int E) {
    int gwarp  = (blockIdx.x * blockDim.x + threadIdx.x) >> 5;
    int lane   = threadIdx.x & 31;
    int nwarps = (gridDim.x * blockDim.x) >> 5;
    for (int e = gwarp; e < E; e += nwarps) {
        int en = e + nwarps;
        if (en < E)
            __builtin_prefetch(X + (size_t)src[en] * KD + coloff + lane * 4, 0, 0);
        float w = norm[e];
        int s = src[e], d = dst[e];
        v4bf x = *(const v4bf*)(X + (size_t)s * KD + coloff + lane * 4);
        float* op = outp + (size_t)d * CH + lane * 4;
        atomicAdd(op + 0, w * (float)x[0]);
        atomicAdd(op + 1, w * (float)x[1]);
        atomicAdd(op + 2, w * (float)x[2]);
        atomicAdd(op + 3, w * (float)x[3]);
    }
}

// ----------------------------------------------------------- pool GEMM (WMMA)
// Xp = X[t] (N x 128 fp32) @ Bp (128x128 col-major bf16) -> bf16 into A[:,0:128]
__global__ __launch_bounds__(256)
void k_pool_gemm(const float* __restrict__ Xt, const bf16_t* __restrict__ Bp,
                 bf16_t* __restrict__ A, int M) {
    int wave = threadIdx.x >> 5;
    int lane = threadIdx.x & 31;
    int l16 = lane & 15, hi = lane >> 4;
    int mbase = blockIdx.x * 16;
    if (mbase >= M) return;
    int row = mbase + l16;
    int col = wave * 16 + l16;
    v8f acc = {};
    #pragma unroll
    for (int kc = 0; kc < 128; kc += 32) {
        const float* ap = Xt + (size_t)row * FEAT + kc + hi * 16;
        f32x8 lo = *(const f32x8*)(ap);
        f32x8 hh = *(const f32x8*)(ap + 8);
        v16bf a;
        #pragma unroll
        for (int j = 0; j < 8; ++j) { a[j] = (bf16_t)lo[j]; a[j + 8] = (bf16_t)hh[j]; }
        v16bf b = *(const v16bf*)(Bp + (size_t)col * 128 + kc + hi * 16);
        acc = __builtin_amdgcn_wmma_f32_16x16x32_bf16(false, a, false, b,
                                                      (short)0, acc, false, false);
    }
    #pragma unroll
    for (int v = 0; v < 8; ++v)
        A[(size_t)(mbase + v + hi * 8) * KD + col] = (bf16_t)acc[v];
}

// ------------------------------------------ fused gate GEMM + LSTM epilogue
// Per block: 16-row stripe x all 512 gate columns. Accumulators -> LDS ->
// in-block gate math -> Cst, H, out[:,t,:]. No global pre-activation buffer.
__global__ __launch_bounds__(256)
void k_gemm_gates(const bf16_t* __restrict__ A, const bf16_t* __restrict__ Bt,
                  const float* __restrict__ bias512,
                  float* __restrict__ Cst, float* __restrict__ H,
                  const float* __restrict__ pw, const float* __restrict__ gb,
                  float* __restrict__ out, int t, int M) {
    __shared__ float Gs[16 * NG];   // 32 KB of the 320 KB WGP LDS
    int wave = threadIdx.x >> 5;
    int lane = threadIdx.x & 31;
    int l16 = lane & 15, hi = lane >> 4;
    int mbase = blockIdx.x * 16;
    if (mbase >= M) return;
    int row = mbase + l16;
    int nbase = wave * 64;

    v8f acc[4];
    #pragma unroll
    for (int nt = 0; nt < 4; ++nt) {
        float bv = bias512[nbase + nt * 16 + l16];
        #pragma unroll
        for (int v = 0; v < 8; ++v) acc[nt][v] = bv;
    }
    for (int kc = 0; kc < KD; kc += 32) {
        const bf16_t* ap = A + (size_t)row * KD + kc + hi * 16;
        v8bf alo = *(const v8bf*)(ap);
        v8bf ahi = *(const v8bf*)(ap + 8);
        v16bf a;
        #pragma unroll
        for (int j = 0; j < 8; ++j) { a[j] = alo[j]; a[j + 8] = ahi[j]; }
        #pragma unroll
        for (int nt = 0; nt < 4; ++nt) {
            int col = nbase + nt * 16 + l16;
            v16bf b = *(const v16bf*)(Bt + (size_t)col * KD + kc + hi * 16);
            acc[nt] = __builtin_amdgcn_wmma_f32_16x16x32_bf16(false, a, false, b,
                                                              (short)0, acc[nt], false, false);
        }
    }
    #pragma unroll
    for (int nt = 0; nt < 4; ++nt)
        #pragma unroll
        for (int v = 0; v < 8; ++v)
            Gs[(v + hi * 8) * NG + nbase + nt * 16 + l16] = acc[nt][v];
    __syncthreads();

    // 16 rows x 128 channels = 2048 gate updates; 8 per thread
    for (int i = threadIdx.x; i < 16 * CH; i += 256) {
        int r = i >> 7, c = i & 127;
        int n = mbase + r;
        float gi = Gs[r * NG + c];
        float gf = Gs[r * NG + 128 + c];
        float gc = Gs[r * NG + 256 + c];
        float go = Gs[r * NG + 384 + c];
        size_t nc = (size_t)n * CH + c;
        float Cold = Cst[nc];
        float I  = sigf(gi + pw[c]       * Cold + gb[c]);
        float Fg = sigf(gf + pw[128 + c] * Cold + gb[128 + c]);
        float Tc = tanhf(gc + gb[256 + c]);
        float Cn = Fg * Cold + I * Tc;
        float O  = sigf(go + pw[256 + c] * Cn + gb[384 + c]);
        float h  = O * tanhf(Cn);
        Cst[nc] = Cn;
        H[nc]   = h;
        out[((size_t)n * T_LEN + t) * CH + c] = h;
    }
}

// ---------------------------------------------------------------------------
extern "C" void kernel_launch(void* const* d_in, const int* in_sizes, int n_in,
                              void* d_out, int out_size, void* d_ws, size_t ws_size,
                              hipStream_t stream) {
    const float* X      = (const float*)d_in[0];   // [T, N, 128]
    const int*   ei     = (const int*)d_in[1];     // [2, E]
    const float* W_pool = (const float*)d_in[2];   // [128, 128]
    const float* convW  = (const float*)d_in[3];   // [8, 3, 128, 128]
    const float* conv_b = (const float*)d_in[4];   // [8, 128]
    const float* peep   = (const float*)d_in[5];   // [3, 128]
    const float* gate_b = (const float*)d_in[6];   // [4, 128]
    float* out = (float*)d_out;                    // [N, T, 128]

    const int N = in_sizes[0] / (T_LEN * FEAT);
    const int E = in_sizes[1] / 2;
    const int* src = ei;
    const int* dst = ei + E;

    // ---- carve workspace (256B aligned chunks)
    char* ws = (char*)d_ws;
    size_t off = 0;
    auto carve = [&](size_t bytes) -> void* {
        void* p = ws + off;
        off += (bytes + 255) & ~(size_t)255;
        return p;
    };
    float*  deg     = (float*)carve((size_t)N * 4);
    float*  dinv    = (float*)carve((size_t)N * 4);
    float*  norm    = (float*)carve((size_t)E * 4);
    float*  Pa      = (float*)carve((size_t)N * CH * 4);
    float*  Pb      = (float*)carve((size_t)N * CH * 4);
    float*  Hbuf    = (float*)carve((size_t)N * CH * 4);
    float*  Cbuf    = (float*)carve((size_t)N * CH * 4);
    bf16_t* Abuf    = (bf16_t*)carve((size_t)N * KD * 2);      // N x 768 basis
    bf16_t* Bt      = (bf16_t*)carve((size_t)NG * KD * 2);     // 512 x 768
    bf16_t* Bp      = (bf16_t*)carve((size_t)FEAT * CH * 2);
    float*  bias512 = (float*)carve((size_t)NG * 4);
    (void)ws_size; (void)n_in; (void)out_size;

    const int NC       = N * CH;
    const int nBlkNC   = (NC + 255) / 256;
    const int nBlkE    = (E + 255) / 256;
    const int nBlkN    = (N + 255) / 256;
    const int nBlkProp = (E * 8 + 255) / 256;   // ~4 edges per warp (grid-stride)
    const int mTiles   = (N + 15) / 16;

    // ---- one-time prep (per launch; deterministic)
    hipMemsetAsync(deg,  0, (size_t)N * 4, stream);
    hipMemsetAsync(Hbuf, 0, (size_t)NC * 4, stream);
    hipMemsetAsync(Cbuf, 0, (size_t)NC * 4, stream);
    k_deg <<<nBlkE, 256, 0, stream>>>(src, deg, E);
    k_dinv<<<nBlkN, 256, 0, stream>>>(deg, dinv, N);
    k_norm<<<nBlkE, 256, 0, stream>>>(src, dst, dinv, norm, E);
    k_prep_w<<<(4 * 3 * 128 * 128 + 255) / 256, 256, 0, stream>>>(convW, Bt);
    k_bias <<<2, 256, 0, stream>>>(conv_b, bias512);
    k_poolw<<<64, 256, 0, stream>>>(W_pool, Bp);

    // ---- time loop
    for (int t = 0; t < T_LEN; ++t) {
        // x-side basis: cols [0,128,256)
        k_pool_gemm<<<mTiles, 256, 0, stream>>>(X + (size_t)t * N * FEAT, Bp, Abuf, N);
        hipMemsetAsync(Pa, 0, (size_t)NC * 4, stream);
        k_prop<<<nBlkProp, 256, 0, stream>>>(Abuf, 0, src, dst, norm, Pa, E);
        k_cvt_slice<<<nBlkNC, 256, 0, stream>>>(Pa, Abuf, 128, NC);
        hipMemsetAsync(Pb, 0, (size_t)NC * 4, stream);
        k_prop<<<nBlkProp, 256, 0, stream>>>(Abuf, 128, src, dst, norm, Pb, E);
        k_tx2<<<nBlkNC, 256, 0, stream>>>(Pb, Abuf, 0, NC);

        // h-side basis: cols [384,512,640)
        k_cvt_slice<<<nBlkNC, 256, 0, stream>>>(Hbuf, Abuf, 384, NC);
        hipMemsetAsync(Pa, 0, (size_t)NC * 4, stream);
        k_prop<<<nBlkProp, 256, 0, stream>>>(Abuf, 384, src, dst, norm, Pa, E);
        k_cvt_slice<<<nBlkNC, 256, 0, stream>>>(Pa, Abuf, 512, NC);
        hipMemsetAsync(Pb, 0, (size_t)NC * 4, stream);
        k_prop<<<nBlkProp, 256, 0, stream>>>(Abuf, 512, src, dst, norm, Pb, E);
        k_tx2<<<nBlkNC, 256, 0, stream>>>(Pb, Abuf, 384, NC);

        // fused 768-K WMMA GEMM + LSTM gates -> Cst, H, out[:,t,:]
        k_gemm_gates<<<mTiles, 256, 0, stream>>>(Abuf, Bt, bias512, Cbuf, Hbuf,
                                                 peep, gate_b, out, t, N);
    }
}
